// SwinBlock_38895223833009
// MI455X (gfx1250) — matrix operand order repaired and checked by reading, not compiled, not run on hardware
//
#include <hip/hip_runtime.h>
#include <hip/hip_bf16.h>

typedef __bf16 bf16;
typedef __attribute__((ext_vector_type(16))) __bf16 v16bf;
typedef __attribute__((ext_vector_type(8)))  float  v8f;
typedef unsigned long long u64;
typedef unsigned int u32;

#define DEVINL __device__ __forceinline__

// ---------------- geometry ----------------
constexpr int BATCH = 32, HPX = 112, WPX = 112, CH = 128, NHEAD = 8;
constexpr int NWIN = 49, NPAD = 64, WSZ = 7, SSZ = 3;
constexpr int NWW = 16;                 // windows per row (112/7)
constexpr int NW_IMG = 256;             // windows per image
constexpr int W_TOT = BATCH * NW_IMG;   // 8192 windows
constexpr int TOK   = BATCH * HPX * WPX; // 401408 tokens
constexpr float QSCALE = 0.25f;          // HD^-0.5, HD=16

// ---------------- workspace layout (bytes); weights stored TRANSPOSED [N][K] ----------------
constexpr size_t XN_OFF  = 0;                                       // xn f32 [TOK][128]
constexpr size_t XW_OFF  = XN_OFF  + (size_t)TOK * CH * 4;          // xw bf16 [8192][64][128]
constexpr size_t WQ_OFF  = XW_OFF  + (size_t)W_TOT * NPAD * CH * 2; // qkv_w^T bf16 [384][128]
constexpr size_t WPJ_OFF = WQ_OFF  + (size_t)CH * 3 * CH * 2;       // proj_w^T bf16 [128][128]
constexpr size_t W1_OFF  = WPJ_OFF + (size_t)CH * CH * 2;           // w1^T bf16 [512][128]
constexpr size_t W2_OFF  = W1_OFF  + (size_t)CH * 4 * CH * 2;       // w2^T bf16 [128][512]
constexpr size_t TAB_OFF = W2_OFF  + (size_t)CH * 4 * CH * 2;       // bias table f32, 512KB
constexpr int TAB_N = 4 * NHEAD * 4 * 4 * 32 * 8;                   // 131072 entries

// ---------------- k_attn LDS layout (bytes), total 131072 < 160K ----------------
constexpr int A_OFF_QK  = 0;       // bf16 [64][256]   = 32768   Q|K   (phase 1-2)
constexpr int A_OFF_VT  = 32768;   // bf16 [128][64]   = 16384   V^T   (phase 1-2)
constexpr int A_OFF_X   = 49152;   // bf16 [64][128]   = 16384         (phase 1)
constexpr int A_OFF_W   = 65536;   // bf16 [384][64]   = 49152         (phase 1, K-half of Wqkv^T)
constexpr int A_OFF_P   = 49152;   // bf16 [8][64][64] = 65536         (phase 2, overlaps X+W)
constexpr int A_OFF_O   = 114688;  // bf16 [64][128]   = 16384         (phase 2-3)
constexpr int A_OFF_WP  = 49152;   // bf16 [128][128]  = 32768         (phase 3, overlaps P)
constexpr int A_LDS     = 131072;

// ---------------- k_mlp LDS layout ----------------
constexpr int M_OFF_H   = 0;       // bf16 [64][128] = 16384
constexpr int M_OFF_HID = 16384;   // bf16 [64][512] = 65536
constexpr int M_OFF_W   = 81920;   // bf16 staging   = 65536
constexpr int M_LDS     = 147456;

// ---------------- WMMA helpers (CDNA5, wave32) ----------------
DEVINL v8f zero8() {
  v8f z;
  #pragma unroll
  for (int i = 0; i < 8; ++i) z[i] = 0.f;
  return z;
}
DEVINL v16bf zero16() {
  v16bf z;
  #pragma unroll
  for (int i = 0; i < 16; ++i) z[i] = (bf16)0.f;
  return z;
}

DEVINL v8f wmma_bf16(v16bf a, v16bf b, v8f c) {
  // v_wmma_f32_16x16x32_bf16  D = A(16x32) * B(32x16) + C
  return __builtin_amdgcn_wmma_f32_16x16x32_bf16(false, a, false, b, (short)0, c, false, false);
}

// A fragment (16x32 bf16) from row-major [M][ldm] LDS tile (one contiguous 32B run).
DEVINL v16bf load_a(const bf16* p, int ldm, int mbase, int kbase) {
  const int lane = threadIdx.x & 31;
  const bf16* row = p + (size_t)(mbase + (lane & 15)) * ldm + kbase + (lane >> 4) * 16;
  v16bf a;
  #pragma unroll
  for (int e = 0; e < 16; ++e) a[e] = row[e];
  return a;
}

// B fragment (32x16 bf16) from TRANSPOSED [N][ldk] LDS tile (one contiguous 32B run).
DEVINL v16bf load_bt(const bf16* p, int ldk, int nbase, int kbase) {
  const int lane = threadIdx.x & 31;
  const bf16* row = p + (size_t)(nbase + (lane & 15)) * ldk + kbase + (lane >> 4) * 16;
  v16bf b;
  #pragma unroll
  for (int e = 0; e < 16; ++e) b[e] = row[e];
  return b;
}

DEVINL int region1(int v) { return (v < 105) ? 0 : ((v < 109) ? 1 : 2); }

// ================= K0: fp32 -> bf16 weight conversion, transposed to [N][K] =================
__global__ __launch_bounds__(256) void k_cvt(const float* __restrict__ qkv_w,
                                             const float* __restrict__ proj_w,
                                             const float* __restrict__ w1,
                                             const float* __restrict__ w2,
                                             bf16* dq, bf16* dp, bf16* d1, bf16* d2) {
  int i = blockIdx.x * 256 + threadIdx.x;
  if (i < 49152) {                                  // qkv_w [128][384] -> [384][128]
    int k = i / 384, n = i % 384;
    dq[(size_t)n * 128 + k] = (bf16)qkv_w[i];
  } else if (i < 65536) {                           // proj_w [128][128] -> transposed
    int j = i - 49152; int k = j / 128, n = j % 128;
    dp[(size_t)n * 128 + k] = (bf16)proj_w[j];
  } else if (i < 131072) {                          // w1 [128][512] -> [512][128]
    int j = i - 65536; int k = j / 512, n = j % 512;
    d1[(size_t)n * 128 + k] = (bf16)w1[j];
  } else if (i < 196608) {                          // w2 [512][128] -> [128][512]
    int j = i - 131072; int k = j / 128, n = j % 128;
    d2[(size_t)n * 512 + k] = (bf16)w2[j];
  }
}

// ================= K0c: fused rel-pos bias + shift mask + pad mask table =================
// Layout: [wt(4)][head(8)][mt(4)][nt(4)][lane(32)][e(8)] f32 — exact WMMA D-fragment order.
// wt bit1: window in last row band (wh==15); wt bit0: last col band (ww==15).
__global__ __launch_bounds__(256) void k_bias(const float* __restrict__ rpb,
                                              float* __restrict__ tab) {
  const int i = blockIdx.x * 256 + threadIdx.x;               // < 131072
  const int e = i & 7, lane = (i >> 3) & 31, nt = (i >> 8) & 3;
  const int mt = (i >> 10) & 3, head = (i >> 12) & 7, wt = (i >> 15) & 3;
  const int n = mt * 16 + (lane >> 4) * 8 + e;                // query token (row)
  const int m = nt * 16 + (lane & 15);                        // key token (col)
  float v;
  if (m >= NWIN) {
    v = -1e30f;                                               // kill padded key columns
  } else {
    const int r2 = m / 7, c2 = m % 7;
    const int r1 = (n < NWIN) ? n / 7 : 0, c1 = (n < NWIN) ? n % 7 : 0;
    v = rpb[((r1 - r2 + 6) * 13 + (c1 - c2 + 6)) * NHEAD + head];
    const int rregH = (wt & 2) ? region1(105 + r1) : 0;
    const int rregW = (wt & 1) ? region1(105 + c1) : 0;
    const int cregH = (wt & 2) ? region1(105 + r2) : 0;
    const int cregW = (wt & 1) ? region1(105 + c2) : 0;
    if (rregH * 3 + rregW != cregH * 3 + cregW) v -= 100.f;
  }
  tab[i] = v;
}

// ================= K0b: zero padding rows 49..63 of xw =================
__global__ __launch_bounds__(256) void k_zeropad(bf16* xw) {
  int i = blockIdx.x * 256 + threadIdx.x;   // one u32 (2 bf16) each
  int widx = i / 960, j = i - widx * 960;   // 15 rows * 128 ch = 1920 bf16 = 960 u32
  if (widx < W_TOT)
    ((u32*)(xw + (size_t)widx * NPAD * CH + NWIN * CH))[j] = 0u;
}

// ================= K1: LN1 + shift + window-partition =================
__global__ __launch_bounds__(256) void k_ln1(const float* __restrict__ x,
                                             const float* __restrict__ lnw,
                                             const float* __restrict__ lnb,
                                             float* __restrict__ xn,
                                             bf16* __restrict__ xw) {
  const int lane = threadIdx.x & 31, wave = threadIdx.x >> 5;
  const int tok = blockIdx.x * 8 + wave;                 // one wave32 per token
  const float4 v = *(const float4*)(x + (size_t)tok * CH + lane * 4);
  float s  = v.x + v.y + v.z + v.w;
  float ss = v.x * v.x + v.y * v.y + v.z * v.z + v.w * v.w;
  #pragma unroll
  for (int d = 16; d >= 1; d >>= 1) { s += __shfl_xor(s, d, 32); ss += __shfl_xor(ss, d, 32); }
  const float mu = s * (1.f / 128.f);
  const float rstd = rsqrtf(ss * (1.f / 128.f) - mu * mu + 1e-5f);
  const float4 wv = *(const float4*)(lnw + lane * 4);
  const float4 bv = *(const float4*)(lnb + lane * 4);
  float4 o;
  o.x = (v.x - mu) * rstd * wv.x + bv.x;  o.y = (v.y - mu) * rstd * wv.y + bv.y;
  o.z = (v.z - mu) * rstd * wv.z + bv.z;  o.w = (v.w - mu) * rstd * wv.w + bv.w;
  *(float4*)(xn + (size_t)tok * CH + lane * 4) = o;

  // shifted window coords: shifted[h'] = xn[h'+3] -> token (h,w) lands at h'=(h-3)%112
  const int bimg = tok / (HPX * WPX);
  const int rem = tok - bimg * HPX * WPX;
  const int h = rem / WPX, w = rem - (rem / WPX) * WPX;
  int hs = h - SSZ; if (hs < 0) hs += HPX;
  int wss = w - SSZ; if (wss < 0) wss += WPX;
  const int widx = bimg * NW_IMG + (hs / WSZ) * NWW + (wss / WSZ);
  const int wrow = (hs % WSZ) * WSZ + (wss % WSZ);
  union { bf16 h4[4]; u64 u; } pk;
  pk.h4[0] = (bf16)o.x; pk.h4[1] = (bf16)o.y; pk.h4[2] = (bf16)o.z; pk.h4[3] = (bf16)o.w;
  *(u64*)(xw + ((size_t)widx * NPAD + wrow) * CH + lane * 4) = pk.u;
}

// ================= K2: fused QKV + attention + proj, one block per window =================
__global__ __launch_bounds__(256) void k_attn(const bf16* __restrict__ xw,
                                              const bf16* __restrict__ wqkv,   // [384][128] ^T
                                              const bf16* __restrict__ wproj,  // [128][128] ^T
                                              const float* __restrict__ qkv_b,
                                              const float* __restrict__ proj_b,
                                              const float* __restrict__ btab,  // fused bias table
                                              const float* __restrict__ xn,
                                              float* __restrict__ y) {
  extern __shared__ char smem[];
  bf16*  sQK  = (bf16*)(smem + A_OFF_QK);   // [64][256]  Q|K
  bf16*  sVt  = (bf16*)(smem + A_OFF_VT);   // [128][64]  V^T (ch-major)
  bf16*  sX   = (bf16*)(smem + A_OFF_X);
  bf16*  sW   = (bf16*)(smem + A_OFF_W);    // [384][64]  K-half of Wqkv^T
  bf16*  sP   = (bf16*)(smem + A_OFF_P);
  bf16*  sO   = (bf16*)(smem + A_OFF_O);

  const int tid = threadIdx.x, lane = tid & 31, wv = tid >> 5;
  const int lhalf = lane >> 4, l16 = lane & 15;
  const int widx = blockIdx.x;
  const int bimg = widx >> 8, iw = widx & 255, wh = iw >> 4, ww = iw & 15;

  // -------- phase 1: QKV = X[64,128] @ Wqkv[128,384] (+bias, Q*scale) --------
  {
    v8f acc[4][3];
    #pragma unroll
    for (int mt = 0; mt < 4; ++mt)
      #pragma unroll
      for (int nt = 0; nt < 3; ++nt) acc[mt][nt] = zero8();

    #pragma unroll 1
    for (int kh = 0; kh < 2; ++kh) {
      { // stage K-half of Wqkv^T: sW[n][0..63] = wqkv[n][kh*64 + 0..63]  (3072 uint4)
        #pragma unroll
        for (int i = 0; i < 12; ++i) {
          const int idx = tid + 256 * i, n = idx >> 3, c8 = idx & 7;
          ((uint4*)sW)[idx] = *(const uint4*)(wqkv + (size_t)n * 128 + kh * 64 + c8 * 8);
        }
      }
      if (kh == 0) {
        const uint4* src = (const uint4*)(xw + (size_t)widx * NPAD * CH);
        #pragma unroll
        for (int i = 0; i < 4; ++i) ((uint4*)sX)[tid + 256 * i] = src[tid + 256 * i];
      }
      __syncthreads();
      #pragma unroll
      for (int mt = 0; mt < 4; ++mt) {
        #pragma unroll
        for (int k2 = 0; k2 < 2; ++k2) {
          v16bf a = load_a(sX, CH, mt * 16, kh * 64 + k2 * 32);
          #pragma unroll
          for (int nt = 0; nt < 3; ++nt) {
            v16bf b = load_bt(sW, 64, wv * 48 + nt * 16, k2 * 32);
            acc[mt][nt] = wmma_bf16(a, b, acc[mt][nt]);
          }
        }
      }
      __syncthreads();
    }
    // store Q|K to sQK (ld 256), V transposed to sVt[ch][tok]
    #pragma unroll
    for (int nt = 0; nt < 3; ++nt) {
      const int colbase = wv * 48 + nt * 16;
      const int col = colbase + l16;
      const float bb = qkv_b[col];
      if (colbase < 128) {            // Q (scaled)
        #pragma unroll
        for (int mt = 0; mt < 4; ++mt)
          #pragma unroll
          for (int e = 0; e < 8; ++e)
            sQK[(size_t)(mt * 16 + lhalf * 8 + e) * 256 + col] = (bf16)((acc[mt][nt][e] + bb) * QSCALE);
      } else if (colbase < 256) {     // K
        #pragma unroll
        for (int mt = 0; mt < 4; ++mt)
          #pragma unroll
          for (int e = 0; e < 8; ++e)
            sQK[(size_t)(mt * 16 + lhalf * 8 + e) * 256 + col] = (bf16)(acc[mt][nt][e] + bb);
      } else {                        // V -> transposed (rows contiguous per lane -> b128 stores)
        #pragma unroll
        for (int mt = 0; mt < 4; ++mt)
          #pragma unroll
          for (int e = 0; e < 8; ++e)
            sVt[(size_t)(col - 256) * 64 + mt * 16 + lhalf * 8 + e] = (bf16)(acc[mt][nt][e] + bb);
      }
    }
  }
  __syncthreads();

  // -------- phase 2: per-wave = per-head attention --------
  {
    const int head = wv;
    bf16* myP = sP + head * 64 * 64;

    // Q as A fragments, HD=16 zero-padded to K=32: lanes with lhalf=1 carry K=16..31 -> zero pad
    v16bf qf[4];
    #pragma unroll
    for (int mt = 0; mt < 4; ++mt) {
      const bf16* row = sQK + (size_t)(mt * 16 + l16) * 256 + head * 16;
      v16bf a = zero16();
      if (!lhalf) {
        #pragma unroll
        for (int e = 0; e < 16; ++e) a[e] = row[e];
      }
      qf[mt] = a;
    }
    // K^T as B fragments: lanes with lhalf=1 carry K=16..31 -> zero pad (exec-masked 32B load)
    v16bf kf[4];
    #pragma unroll
    for (int nt = 0; nt < 4; ++nt) {
      v16bf b = zero16();
      if (!lhalf) {
        const bf16* src = sQK + (size_t)(nt * 16 + l16) * 256 + 128 + head * 16;
        #pragma unroll
        for (int e = 0; e < 16; ++e) b[e] = src[e];
      }
      kf[nt] = b;
    }
    v8f S[4][4];
    #pragma unroll
    for (int mt = 0; mt < 4; ++mt)
      #pragma unroll
      for (int nt = 0; nt < 4; ++nt) S[mt][nt] = wmma_bf16(qf[mt], kf[nt], zero8());

    // fused rel-pos bias + shift mask + pad mask: precomputed in fragment order
    {
      const int wt = ((wh == 15) ? 2 : 0) | ((ww == 15) ? 1 : 0);
      const float* tb = btab + (((size_t)wt * NHEAD + head) * 16 * 32 * 8) + (size_t)lane * 8;
      #pragma unroll
      for (int mt = 0; mt < 4; ++mt)
        #pragma unroll
        for (int nt = 0; nt < 4; ++nt) {
          const float* src = tb + (size_t)(mt * 4 + nt) * 32 * 8;   // 32B contiguous per lane
          #pragma unroll
          for (int e = 0; e < 8; ++e) S[mt][nt][e] += src[e];
        }
    }
    // row softmax: row lives in one 16-lane half across 4 col-tiles -> xor-reduce masks 8..1
    #pragma unroll
    for (int mt = 0; mt < 4; ++mt)
      #pragma unroll
      for (int e = 0; e < 8; ++e) {
        float mx = fmaxf(fmaxf(S[mt][0][e], S[mt][1][e]), fmaxf(S[mt][2][e], S[mt][3][e]));
        #pragma unroll
        for (int d = 8; d >= 1; d >>= 1) mx = fmaxf(mx, __shfl_xor(mx, d, 32));
        float sum = 0.f;
        #pragma unroll
        for (int nt = 0; nt < 4; ++nt) { float ex = __expf(S[mt][nt][e] - mx); S[mt][nt][e] = ex; sum += ex; }
        #pragma unroll
        for (int d = 8; d >= 1; d >>= 1) sum += __shfl_xor(sum, d, 32);
        const float inv = 1.f / sum;
        #pragma unroll
        for (int nt = 0; nt < 4; ++nt) S[mt][nt][e] *= inv;
      }
    // bounce P through LDS to re-fragment as A
    #pragma unroll
    for (int mt = 0; mt < 4; ++mt)
      #pragma unroll
      for (int nt = 0; nt < 4; ++nt)
        #pragma unroll
        for (int e = 0; e < 8; ++e)
          myP[(size_t)(mt * 16 + lhalf * 8 + e) * 64 + nt * 16 + l16] = (bf16)S[mt][nt][e];

    // V as B fragments from V^T: contiguous 32B per lane
    v16bf vf[2];
    #pragma unroll
    for (int kk = 0; kk < 2; ++kk) {
      const bf16* src = sVt + (size_t)(head * 16 + l16) * 64 + kk * 32 + lhalf * 16;
      v16bf b;
      #pragma unroll
      for (int e = 0; e < 16; ++e) b[e] = src[e];
      vf[kk] = b;
    }
    #pragma unroll
    for (int mt = 0; mt < 4; ++mt) {
      v8f o = zero8();
      #pragma unroll
      for (int kk = 0; kk < 2; ++kk)
        o = wmma_bf16(load_a(myP, 64, mt * 16, kk * 32), vf[kk], o);
      #pragma unroll
      for (int e = 0; e < 8; ++e)
        sO[(size_t)(mt * 16 + lhalf * 8 + e) * CH + head * 16 + l16] = (bf16)o[e];
    }
  }
  __syncthreads();

  // -------- phase 3: proj [64,128]@[128,128] + un-shift + residual(xn) --------
  {
    const uint4* src = (const uint4*)wproj;   // wproj^T contiguous
    #pragma unroll
    for (int i = 0; i < 8; ++i) ((uint4*)(smem + A_OFF_WP))[tid + 256 * i] = src[tid + 256 * i];
  }
  __syncthreads();
  {
    const bf16* sWp = (const bf16*)(smem + A_OFF_WP);   // [128][128] ^T
    const int nb = wv * 16, col = nb + l16;
    const float pb = proj_b[col];
    #pragma unroll
    for (int mt = 0; mt < 4; ++mt) {
      v8f acc = zero8();
      #pragma unroll
      for (int k = 0; k < 4; ++k)
        acc = wmma_bf16(load_a(sO, CH, mt * 16, k * 32), load_bt(sWp, CH, nb, k * 32), acc);
      #pragma unroll
      for (int e = 0; e < 8; ++e) {
        const int t = mt * 16 + lhalf * 8 + e;
        if (t < NWIN) {
          const int r = t / 7, c = t % 7;
          int hf = wh * 7 + r + SSZ; if (hf >= HPX) hf -= HPX;   // roll back +3
          int wf = ww * 7 + c + SSZ; if (wf >= WPX) wf -= WPX;
          const size_t gi = (((size_t)bimg * HPX + hf) * WPX + wf) * CH + col;
          y[gi] = xn[gi] + acc[e] + pb;
        }
      }
    }
  }
}

// ================= K3: fused LN2 + MLP + residual, 64 tokens per block =================
__global__ __launch_bounds__(256) void k_mlp(const float* __restrict__ y,
                                             const float* __restrict__ lnw,
                                             const float* __restrict__ lnb,
                                             const bf16* __restrict__ w1,   // [512][128] ^T
                                             const float* __restrict__ b1,
                                             const bf16* __restrict__ w2,   // [128][512] ^T
                                             const float* __restrict__ b2,
                                             float* __restrict__ out) {
  extern __shared__ char smem[];
  bf16* sH   = (bf16*)(smem + M_OFF_H);
  bf16* sHid = (bf16*)(smem + M_OFF_HID);
  bf16* sW   = (bf16*)(smem + M_OFF_W);
  const int tid = threadIdx.x, lane = tid & 31, wv = tid >> 5;
  const int lhalf = lane >> 4, l16 = lane & 15;
  const size_t t0 = (size_t)blockIdx.x * 64;

  // ---- LN2: 4 threads per token, 32 channels each ----
  {
    const int row = tid >> 2, q = tid & 3;
    const float* src = y + (t0 + row) * CH + q * 32;
    float vals[32]; float s = 0.f, ss = 0.f;
    #pragma unroll
    for (int i = 0; i < 8; ++i) {
      float4 v = *(const float4*)(src + i * 4);
      vals[i*4+0]=v.x; vals[i*4+1]=v.y; vals[i*4+2]=v.z; vals[i*4+3]=v.w;
      s += v.x+v.y+v.z+v.w; ss += v.x*v.x+v.y*v.y+v.z*v.z+v.w*v.w;
    }
    s  += __shfl_xor(s, 1, 32);  s  += __shfl_xor(s, 2, 32);
    ss += __shfl_xor(ss, 1, 32); ss += __shfl_xor(ss, 2, 32);
    const float mu = s * (1.f / 128.f);
    const float rstd = rsqrtf(ss * (1.f / 128.f) - mu * mu + 1e-5f);
    #pragma unroll
    for (int i = 0; i < 32; ++i) {
      const int cch = q * 32 + i;
      sH[(size_t)row * CH + cch] = (bf16)((vals[i] - mu) * rstd * lnw[cch] + lnb[cch]);
    }
  }
  __syncthreads();

  // ---- GEMM1: hid[64,512] = h @ W1 + b1, gelu(tanh) -> bf16 ; two N-halves of W1^T staged ----
  #pragma unroll 1
  for (int nh = 0; nh < 2; ++nh) {
    { // W1^T rows nh*256..+255 are contiguous: straight 64KB copy
      const uint4* src = (const uint4*)(w1 + (size_t)nh * 256 * 128);
      #pragma unroll
      for (int i = 0; i < 16; ++i) ((uint4*)sW)[tid + 256 * i] = src[tid + 256 * i];
    }
    __syncthreads();
    #pragma unroll
    for (int nt2 = 0; nt2 < 2; ++nt2) {
      const int nloc = (wv * 2 + nt2) * 16;        // local N base within staged half
      const int gcol = nh * 256 + nloc + l16;
      const float bb = b1[gcol];
      #pragma unroll
      for (int mt = 0; mt < 4; ++mt) {
        v8f acc = zero8();
        #pragma unroll
        for (int k = 0; k < 4; ++k)
          acc = wmma_bf16(load_a(sH, CH, mt * 16, k * 32), load_bt(sW, CH, nloc, k * 32), acc);
        #pragma unroll
        for (int e = 0; e < 8; ++e) {
          const int row = mt * 16 + lhalf * 8 + e;
          const float xv = acc[e] + bb;
          const float g = 0.5f * xv * (1.f + tanhf(0.7978845608f * (xv + 0.044715f * xv * xv * xv)));
          sHid[(size_t)row * 512 + gcol] = (bf16)g;
        }
      }
    }
    __syncthreads();
  }

  // ---- GEMM2: out[64,128] = hid @ W2 + b2 + y ; two K-halves of W2^T staged ----
  {
    v8f acc[4];
    #pragma unroll
    for (int mt = 0; mt < 4; ++mt) acc[mt] = zero8();
    const int nb = wv * 16;
    #pragma unroll 1
    for (int kh = 0; kh < 2; ++kh) {
      { // stage sW[n][0..255] = w2t[n][kh*256 + 0..255]  (4096 uint4, strided rows)
        #pragma unroll
        for (int i = 0; i < 16; ++i) {
          const int idx = tid + 256 * i, n = idx >> 5, c8 = idx & 31;
          ((uint4*)sW)[idx] = *(const uint4*)(w2 + (size_t)n * 512 + kh * 256 + c8 * 8);
        }
      }
      __syncthreads();
      #pragma unroll
      for (int mt = 0; mt < 4; ++mt)
        #pragma unroll
        for (int k = 0; k < 8; ++k)
          acc[mt] = wmma_bf16(load_a(sHid, 512, mt * 16, kh * 256 + k * 32),
                              load_bt(sW, 256, nb, k * 32), acc[mt]);
      __syncthreads();
    }
    const int col = nb + l16;
    const float bb = b2[col];
    #pragma unroll
    for (int mt = 0; mt < 4; ++mt)
      #pragma unroll
      for (int e = 0; e < 8; ++e) {
        const int row = mt * 16 + lhalf * 8 + e;
        const size_t gi = (t0 + row) * CH + col;
        out[gi] = y[gi] + acc[mt][e] + bb;   // in-place safe: each element owned by one thread
      }
  }
}

// ================= host launch =================
extern "C" void kernel_launch(void* const* d_in, const int* in_sizes, int n_in,
                              void* d_out, int out_size, void* d_ws, size_t ws_size,
                              hipStream_t stream) {
  const float* x      = (const float*)d_in[0];
  const float* qkv_w  = (const float*)d_in[1];
  const float* qkv_b  = (const float*)d_in[2];
  const float* proj_w = (const float*)d_in[3];
  const float* proj_b = (const float*)d_in[4];
  const float* rpb    = (const float*)d_in[5];
  const float* ln1w   = (const float*)d_in[6];
  const float* ln1b   = (const float*)d_in[7];
  const float* ln2w   = (const float*)d_in[8];
  const float* ln2b   = (const float*)d_in[9];
  const float* w1     = (const float*)d_in[10];
  const float* b1     = (const float*)d_in[11];
  const float* w2     = (const float*)d_in[12];
  const float* b2     = (const float*)d_in[13];

  char* ws = (char*)d_ws;
  float* xn   = (float*)(ws + XN_OFF);
  bf16*  xw   = (bf16*)(ws + XW_OFF);
  bf16*  wq   = (bf16*)(ws + WQ_OFF);
  bf16*  wp   = (bf16*)(ws + WPJ_OFF);
  bf16*  w1b  = (bf16*)(ws + W1_OFF);
  bf16*  w2b  = (bf16*)(ws + W2_OFF);
  float* btab = (float*)(ws + TAB_OFF);
  float* y    = (float*)d_out;         // y lives in d_out; k_mlp updates it in place

  k_cvt<<<768, 256, 0, stream>>>(qkv_w, proj_w, w1, w2, wq, wp, w1b, w2b);
  k_bias<<<TAB_N / 256, 256, 0, stream>>>(rpb, btab);
  k_zeropad<<<(W_TOT * 960) / 256, 256, 0, stream>>>(xw);
  k_ln1<<<TOK / 8, 256, 0, stream>>>(x, ln1w, ln1b, xn, xw);
  k_attn<<<W_TOT, 256, A_LDS, stream>>>(xw, wq, wp, qkv_b, proj_b, btab, xn, y);
  k_mlp<<<TOK / 64, 256, M_LDS, stream>>>(y, ln2w, ln2b, w1b, b1, w2b, b2, y);
}
